// LeapRNNLayer_57217554317374
// MI455X (gfx1250) — matrix operand order repaired and checked by reading, not compiled, not run on hardware
//
#include <hip/hip_runtime.h>
#include <hip/hip_bf16.h>
#include <stdint.h>

// Shapes fixed by the reference: B=32, T=512, D=512, H=512, K=64, 3H=1536.
#define B_   32
#define T_   512
#define D_   512
#define H_   512
#define K_   64
#define G3H  1536
#define KD   (K_ * D_)        // 32768 (im2col K-dimension)
#define EPSG 1e-20f

typedef __attribute__((ext_vector_type(16))) __bf16 v16bf;
typedef __attribute__((ext_vector_type(8)))  float  v8f;
typedef __attribute__((ext_vector_type(4)))  unsigned int v4u;
typedef int v4i_ __attribute__((vector_size(16)));   // matches async builtin param type

union Frag16 {            // one WMMA 16-bit operand: 16 bf16 = 8 VGPRs
    v4u   q[2];           // two 16-byte loads
    v16bf v;
};

__device__ __forceinline__ unsigned short f2bf(float f) {
    union { float f; unsigned int u; } c; c.f = f;
    unsigned int r = c.u + 0x7FFFu + ((c.u >> 16) & 1u);   // round-to-nearest-even
    return (unsigned short)(r >> 16);
}

// ---------------- CDNA5 async global->LDS copy (one-hop, tracked by ASYNCcnt) ----------------

#if __has_builtin(__builtin_amdgcn_global_load_async_to_lds_b128)
#define HAS_ASYNC_LDS 1
#else
#define HAS_ASYNC_LDS 0
#endif

__device__ __forceinline__ void async_copy16(const unsigned short* __restrict__ g,
                                             unsigned short* __restrict__ l) {
#if HAS_ASYNC_LDS
    __builtin_amdgcn_global_load_async_to_lds_b128(
        (__attribute__((address_space(1))) v4i_*)(uintptr_t)g,
        (__attribute__((address_space(3))) v4i_*)(unsigned int)(uintptr_t)l,
        0, 0);
#else
    *(v4u*)l = *(const v4u*)g;
#endif
}

__device__ __forceinline__ void async_drain() {
#if HAS_ASYNC_LDS
#if __has_builtin(__builtin_amdgcn_s_wait_asynccnt)
    __builtin_amdgcn_s_wait_asynccnt(0);
#else
    asm volatile("s_wait_asynccnt 0" ::: "memory");
#endif
#endif
}

// ---------------- converts / transposes (bandwidth-trivial) ----------------

__global__ void cvt_copy(const float* __restrict__ in, unsigned short* __restrict__ o, int n) {
    int i = blockIdx.x * 256 + threadIdx.x;
    if (i < n) o[i] = f2bf(in[i]);
}

// f: [rows][cols] f32  ->  o: [cols][rows] bf16   (so B-operand K runs contiguously)
__global__ void cvt_wT(const float* __restrict__ f, unsigned short* __restrict__ o,
                       int rows, int cols) {
    int i = blockIdx.x * 256 + threadIdx.x;
    if (i >= rows * cols) return;
    int c = i % cols, r = i / cols;
    o[(size_t)c * rows + r] = f2bf(f[i]);
}

__global__ void init_zpad(unsigned short* z) {     // 128-byte zero page for SAME padding
    z[threadIdx.x] = 0;
}

// ================= LDS-tiled, double-buffered, async-pipelined conv GEMM =================
// za[t][b][h] = tanh( sum_{ks,d} x[b, t+ks-31, d] * Wc[h][ks*D+d] )
// Workgroup: 512 threads = 16 waves, tile M=128 (4 consecutive t x 32 b) x N=128.
// Each wave computes 32x32 (2x2 WMMA tiles). K chunk = 64, double-buffered in LDS.
// Pipeline per chunk: drain(cur) -> barrier -> async-issue(nxt) -> 8 WMMAs per wave on cur.
// The drain targets loads issued one iteration earlier, so their latency hides
// under the previous chunk's WMMAs. Out-of-range conv taps read a zero page
// (branchless address select) so every lane always issues its async load.

__global__ __launch_bounds__(512)
void conv_gemm_tiled(const unsigned short* __restrict__ xb,   // [B][T][D] bf16
                     const unsigned short* __restrict__ Wc,   // [H][KD]   bf16
                     const unsigned short* __restrict__ zpad, // 128B zeros
                     float* __restrict__ za)                  // [T][B][H] f32
{
    __shared__ __align__(16) unsigned short As[2][128 * 64];  // [buf][row][k] 2x16KB
    __shared__ __align__(16) unsigned short Bs[2][128 * 64];  // [buf][col][k] 2x16KB

    const int mb   = blockIdx.x >> 2;        // 128 M-blocks (4 t's each)
    const int nb   = blockIdx.x & 3;         // 4 N-blocks of 128 cols
    const int tid  = threadIdx.x;
    const int wv   = tid >> 5;
    const int lane = tid & 31;
    const int wm   = wv >> 2;                // wave row (0..3) == ti
    const int wn   = wv & 3;                 // wave col (0..3)
    const int lm   = lane & 15;
    const bool hi  = lane >= 16;
    const int akb  = hi ? 16 : 0;            // A 16x32 K base (contiguous, matches B)
    const int bkb  = hi ? 16 : 0;            // B 32x16 K base (contiguous, matches A)
    const int t0   = mb * 4;
    const int n0   = nb * 128;

    // This thread's two 16B slots per tile (1024 slots / 512 threads)
    const int row0 = tid >> 3,       c0 = (tid & 7) * 8;          // rep 0
    const int row1 = (512 + tid) >> 3, c1 = ((512 + tid) & 7) * 8; // rep 1

    v8f acc00 = {}, acc01 = {}, acc10 = {}, acc11 = {};

    auto issue = [&](int buf, int ch) {
        const int ks = ch >> 3;              // conv tap (64 | 512: chunk never crosses a tap)
        const int db = (ch & 7) * 64;
        #pragma unroll
        for (int rep = 0; rep < 2; ++rep) {
            const int row = rep ? row1 : row0;
            const int c16 = rep ? c1 : c0;
            // A: row = ti*32 + b ; src x[b][t0+ti+ks-31][db+c16..] or zero page
            const int ti   = row >> 5, b = row & 31;
            const int tsrc = t0 + ti + ks - 31;          // SAME pad: left = 31
            const bool ok  = (tsrc >= 0) && (tsrc < T_);
            const unsigned short* asrc =
                ok ? (xb + ((size_t)b * T_ + tsrc) * D_ + db + c16) : zpad;
            async_copy16(asrc, &As[buf][row * 64 + c16]);
            // B: col-major-in-K slab of Wc
            async_copy16(Wc + (size_t)(n0 + row) * KD + ks * D_ + db + c16,
                         &Bs[buf][row * 64 + c16]);
        }
    };

    issue(0, 0);                             // prologue: fill buffer 0

    for (int ch = 0; ch < KD / 64; ++ch) {   // 512 K-chunks of 64
        const int cur = ch & 1;
        async_drain();                       // own loads for 'cur' landed in LDS
        __syncthreads();                     // all waves' 'cur' data visible;
                                             // all waves done reading buf 'nxt' (prev compute)
        if (ch + 1 < KD / 64) issue(cur ^ 1, ch + 1);   // overlaps with WMMAs below

        const unsigned short* Ac = As[cur];
        const unsigned short* Bc = Bs[cur];
        #pragma unroll
        for (int kk = 0; kk < 64; kk += 32) {
            Frag16 a0, a1, b0, b1;
            const unsigned short* ar0 = Ac + (wm * 32 + lm)      * 64 + kk + akb;
            const unsigned short* ar1 = Ac + (wm * 32 + 16 + lm) * 64 + kk + akb;
            a0.q[0] = *(const v4u*)ar0;  a0.q[1] = *(const v4u*)(ar0 + 8);
            a1.q[0] = *(const v4u*)ar1;  a1.q[1] = *(const v4u*)(ar1 + 8);
            const unsigned short* br0 = Bc + (wn * 32 + lm)      * 64 + kk + bkb;
            const unsigned short* br1 = Bc + (wn * 32 + 16 + lm) * 64 + kk + bkb;
            b0.q[0] = *(const v4u*)br0;  b0.q[1] = *(const v4u*)(br0 + 8);
            b1.q[0] = *(const v4u*)br1;  b1.q[1] = *(const v4u*)(br1 + 8);
            acc00 = __builtin_amdgcn_wmma_f32_16x16x32_bf16(false, a0.v, false, b0.v, (short)0, acc00, false, false);
            acc01 = __builtin_amdgcn_wmma_f32_16x16x32_bf16(false, a0.v, false, b1.v, (short)0, acc01, false, false);
            acc10 = __builtin_amdgcn_wmma_f32_16x16x32_bf16(false, a1.v, false, b0.v, (short)0, acc10, false, false);
            acc11 = __builtin_amdgcn_wmma_f32_16x16x32_bf16(false, a1.v, false, b1.v, (short)0, acc11, false, false);
        }
    }

    // C/D layout: VGPR r -> row r (lanes 0-15) / r+8 (lanes 16-31); col = lane&15
    const int m0   = mb * 128;
    const int rofs = hi ? 8 : 0;
    #pragma unroll
    for (int r = 0; r < 8; ++r) {
        int ro0 = m0 + wm * 32 + r + rofs;
        int ro1 = ro0 + 16;
        int co0 = n0 + wn * 32 + lm;
        int co1 = co0 + 16;
        za[(size_t)ro0 * H_ + co0] = tanhf(acc00[r]);
        za[(size_t)ro0 * H_ + co1] = tanhf(acc01[r]);
        za[(size_t)ro1 * H_ + co0] = tanhf(acc10[r]);
        za[(size_t)ro1 * H_ + co1] = tanhf(acc11[r]);
    }
}

// ================= direct-load WMMA GEMM (used for xm = x @ gru_kernel, 26 GFLOP) =================

template<int NCOLS, int KSL, int OFF, bool DOTANH>
__global__ __launch_bounds__(256)
void gemm_kernel(const unsigned short* __restrict__ xb,   // [B][T][D] bf16
                 const unsigned short* __restrict__ Wt,   // [NCOLS][KSL*D] bf16
                 float* __restrict__ out)                 // [T][B][NCOLS] f32
{
    const int gtid = blockIdx.x * 256 + threadIdx.x;
    const int wave = gtid >> 5;
    const int lane = gtid & 31;
    const int nblocks = NCOLS / 32;
    const int mblk = wave / nblocks;          // == t
    const int nblk = wave % nblocks;
    if (mblk >= T_) return;
    const int t  = mblk;
    const int n0 = nblk * 32;
    const int lm = lane & 15;
    const bool hi = lane >= 16;
    const int akb = hi ? 16 : 0;
    const int bkb = hi ? 16 : 0;

    v8f acc00 = {}, acc01 = {}, acc10 = {}, acc11 = {};

    for (int ks = 0; ks < KSL; ++ks) {
        const int tsrc = t + ks - OFF;
        if (tsrc < 0 || tsrc >= T_) continue;
        const unsigned short* arow0 = xb + ((size_t)(lm)      * T_ + tsrc) * D_;
        const unsigned short* arow1 = xb + ((size_t)(16 + lm) * T_ + tsrc) * D_;
        const size_t wrow = (size_t)(KSL * D_);
        for (int db = 0; db < D_; db += 32) {
            Frag16 a0, a1, b0, b1;
            a0.q[0] = *(const v4u*)(arow0 + db + akb);
            a0.q[1] = *(const v4u*)(arow0 + db + akb + 8);
            a1.q[0] = *(const v4u*)(arow1 + db + akb);
            a1.q[1] = *(const v4u*)(arow1 + db + akb + 8);
            const unsigned short* br0 = Wt + (size_t)(n0 + lm)      * wrow + ks * D_ + db + bkb;
            const unsigned short* br1 = Wt + (size_t)(n0 + 16 + lm) * wrow + ks * D_ + db + bkb;
            b0.q[0] = *(const v4u*)(br0);
            b0.q[1] = *(const v4u*)(br0 + 8);
            b1.q[0] = *(const v4u*)(br1);
            b1.q[1] = *(const v4u*)(br1 + 8);
            acc00 = __builtin_amdgcn_wmma_f32_16x16x32_bf16(false, a0.v, false, b0.v, (short)0, acc00, false, false);
            acc01 = __builtin_amdgcn_wmma_f32_16x16x32_bf16(false, a0.v, false, b1.v, (short)0, acc01, false, false);
            acc10 = __builtin_amdgcn_wmma_f32_16x16x32_bf16(false, a1.v, false, b0.v, (short)0, acc10, false, false);
            acc11 = __builtin_amdgcn_wmma_f32_16x16x32_bf16(false, a1.v, false, b1.v, (short)0, acc11, false, false);
        }
    }

    const int m0 = t * 32;
    const int rofs = hi ? 8 : 0;
    #pragma unroll
    for (int r = 0; r < 8; ++r) {
        float v00 = acc00[r], v01 = acc01[r], v10 = acc10[r], v11 = acc11[r];
        if (DOTANH) { v00 = tanhf(v00); v01 = tanhf(v01); v10 = tanhf(v10); v11 = tanhf(v11); }
        int b0r = r + rofs, b1r = 16 + r + rofs;
        out[(size_t)(m0 + b0r) * NCOLS + n0 + lm]      = v00;
        out[(size_t)(m0 + b0r) * NCOLS + n0 + 16 + lm] = v01;
        out[(size_t)(m0 + b1r) * NCOLS + n0 + lm]      = v10;
        out[(size_t)(m0 + b1r) * NCOLS + n0 + 16 + lm] = v11;
    }
}

// ---------------- persistent GRU scan: 1 workgroup, 32 waves, one 16-col group per wave ----------

__global__ __launch_bounds__(1024)
void scan_kernel(const float* __restrict__ xm,          // [T][B][3H]  x@gru_kernel
                 const float* __restrict__ za,          // [T][B][H]   tanh(conv)
                 const unsigned short* __restrict__ Wr, // [3H][H] bf16 (transposed rec kernel)
                 const float* __restrict__ f_skip,      // [2H][2]
                 const float* __restrict__ gbias,       // [2][3H]
                 const float* __restrict__ ug,          // [T][B][2]
                 float* __restrict__ out)               // [B][T][H]
{
    __shared__ __align__(16) unsigned short h_b[B_ * H_]; // h in bf16 (WMMA A operand)
    __shared__ float h_f[B_ * H_];                        // h in f32 (gate math, logits)
    __shared__ float logitsS[B_ * 2];
    __shared__ float skipS[B_ * 2];

    const int tid  = threadIdx.x;
    const int w    = tid >> 5;        // wave id == column group (16 cols per gate)
    const int lane = tid & 31;
    const int lm   = lane & 15;
    const bool hi  = lane >= 16;
    const int akb  = hi ? 16 : 0;
    const int bkb  = hi ? 16 : 0;

    for (int i = tid; i < B_ * H_; i += 1024) { h_f[i] = 0.0f; h_b[i] = 0; }
    __syncthreads();

    const float* bin  = gbias;         // input bias  [3H]
    const float* brec = gbias + G3H;   // recurrent bias [3H]

    for (int t = 0; t < T_; ++t) {
        // --- phase A: hm tiles for this wave's columns: acc[i][g] (i = row half, g = z/r/h gate)
        v8f acc00 = {}, acc01 = {}, acc02 = {}, acc10 = {}, acc11 = {}, acc12 = {};
        for (int kb = 0; kb < H_; kb += 32) {
            Frag16 a0, a1, bz, br_, bh;
            a0.q[0] = *(const v4u*)(h_b + (lm)      * H_ + kb + akb);
            a0.q[1] = *(const v4u*)(h_b + (lm)      * H_ + kb + akb + 8);
            a1.q[0] = *(const v4u*)(h_b + (16 + lm) * H_ + kb + akb);
            a1.q[1] = *(const v4u*)(h_b + (16 + lm) * H_ + kb + akb + 8);
            const unsigned short* pz = Wr + (size_t)(0 * H_ + w * 16 + lm) * H_ + kb + bkb;
            const unsigned short* pr = Wr + (size_t)(1 * H_ + w * 16 + lm) * H_ + kb + bkb;
            const unsigned short* ph = Wr + (size_t)(2 * H_ + w * 16 + lm) * H_ + kb + bkb;
            bz.q[0] = *(const v4u*)(pz);  bz.q[1] = *(const v4u*)(pz + 8);
            br_.q[0] = *(const v4u*)(pr); br_.q[1] = *(const v4u*)(pr + 8);
            bh.q[0] = *(const v4u*)(ph);  bh.q[1] = *(const v4u*)(ph + 8);
            acc00 = __builtin_amdgcn_wmma_f32_16x16x32_bf16(false, a0.v, false, bz.v,  (short)0, acc00, false, false);
            acc01 = __builtin_amdgcn_wmma_f32_16x16x32_bf16(false, a0.v, false, br_.v, (short)0, acc01, false, false);
            acc02 = __builtin_amdgcn_wmma_f32_16x16x32_bf16(false, a0.v, false, bh.v,  (short)0, acc02, false, false);
            acc10 = __builtin_amdgcn_wmma_f32_16x16x32_bf16(false, a1.v, false, bz.v,  (short)0, acc10, false, false);
            acc11 = __builtin_amdgcn_wmma_f32_16x16x32_bf16(false, a1.v, false, br_.v, (short)0, acc11, false, false);
            acc12 = __builtin_amdgcn_wmma_f32_16x16x32_bf16(false, a1.v, false, bh.v,  (short)0, acc12, false, false);
        }

        // --- phase B: skip-gate logits for batch row b == w (dot over [h_prev ; za_t], len 1024)
        float s0 = 0.0f, s1 = 0.0f;
        for (int j = lane; j < H_; j += 32) {
            float hv = h_f[w * H_ + j];
            float zv = za[((size_t)t * B_ + w) * H_ + j];
            s0 += hv * f_skip[j * 2 + 0] + zv * f_skip[(H_ + j) * 2 + 0];
            s1 += hv * f_skip[j * 2 + 1] + zv * f_skip[(H_ + j) * 2 + 1];
        }
        #pragma unroll
        for (int off = 16; off; off >>= 1) {
            s0 += __shfl_xor(s0, off, 32);
            s1 += __shfl_xor(s1, off, 32);
        }
        if (lane == 0) { logitsS[w * 2 + 0] = s0; logitsS[w * 2 + 1] = s1; }
        __syncthreads();   // all reads of h done; logits visible

        // --- phase C: gumbel-softmax of logits (one thread per batch row)
        if (tid < B_) {
            float u0 = ug[((size_t)t * B_ + tid) * 2 + 0];
            float u1 = ug[((size_t)t * B_ + tid) * 2 + 1];
            float g0 = -logf(-logf(u0 + EPSG) + EPSG);
            float g1 = -logf(-logf(u1 + EPSG) + EPSG);
            float l0 = logitsS[tid * 2 + 0] + g0;
            float l1 = logitsS[tid * 2 + 1] + g1;
            float mx = fmaxf(l0, l1);
            float e0 = expf(l0 - mx), e1 = expf(l1 - mx);
            float inv = 1.0f / (e0 + e1);
            skipS[tid * 2 + 0] = e0 * inv;
            skipS[tid * 2 + 1] = e1 * inv;
        }
        __syncthreads();   // skipS visible

        // --- phase D: GRU gate math + skip blend for this wave's 16 columns, all 32 rows
        const int col = w * 16 + lm;
        const int rofs = hi ? 8 : 0;
        #pragma unroll
        for (int i = 0; i < 2; ++i) {
            #pragma unroll
            for (int r = 0; r < 8; ++r) {
                int b = i * 16 + r + rofs;
                const float* xrow = xm + ((size_t)t * B_ + b) * G3H;
                float xz = xrow[col]           + bin[col];
                float xr = xrow[H_ + col]      + bin[H_ + col];
                float xh = xrow[2 * H_ + col]  + bin[2 * H_ + col];
                float hz = (i == 0 ? acc00[r] : acc10[r]) + brec[col];
                float hr = (i == 0 ? acc01[r] : acc11[r]) + brec[H_ + col];
                float hh = (i == 0 ? acc02[r] : acc12[r]) + brec[2 * H_ + col];
                float hp = h_f[b * H_ + col];
                float zg = 1.0f / (1.0f + expf(-(xz + hz)));
                float rg = 1.0f / (1.0f + expf(-(xr + hr)));
                float hc = tanhf(xh + rg * hh);
                float ht = zg * hp + (1.0f - zg) * hc;
                ht = skipS[b * 2 + 0] * hp + skipS[b * 2 + 1] * ht;
                h_f[b * H_ + col] = ht;
                h_b[b * H_ + col] = f2bf(ht);
                out[((size_t)b * T_ + t) * H_ + col] = ht;
            }
        }
        __syncthreads();   // writes of h visible before next step's reads
    }
}

// ------------------------------------ host ------------------------------------

extern "C" void kernel_launch(void* const* d_in, const int* in_sizes, int n_in,
                              void* d_out, int out_size, void* d_ws, size_t ws_size,
                              hipStream_t stream)
{
    const float* x       = (const float*)d_in[0];  // [B][T][D]
    const float* f_after = (const float*)d_in[1];  // [K][D][H]
    const float* f_skip  = (const float*)d_in[2];  // [2H][2]
    const float* gru_k   = (const float*)d_in[3];  // [D][3H]
    const float* gru_rk  = (const float*)d_in[4];  // [H][3H]
    const float* gru_b   = (const float*)d_in[5];  // [2][3H]
    const float* ug      = (const float*)d_in[6];  // [T][B][2]
    float* out = (float*)d_out;                    // [B][T][H]

    char* ws = (char*)d_ws;
    unsigned short* xb = (unsigned short*)ws;  ws += (size_t)B_ * T_ * D_ * 2;   // x in bf16
    unsigned short* Wc = (unsigned short*)ws;  ws += (size_t)H_ * KD * 2;        // conv W^T
    unsigned short* Wk = (unsigned short*)ws;  ws += (size_t)G3H * D_ * 2;       // gru_kernel^T
    unsigned short* Wr = (unsigned short*)ws;  ws += (size_t)G3H * H_ * 2;       // gru_rec^T
    float* za = (float*)ws;                    ws += (size_t)T_ * B_ * H_ * 4;   // tanh(conv)
    float* xm = (float*)ws;                    ws += (size_t)T_ * B_ * G3H * 4;  // x@gru_kernel
    unsigned short* zp = (unsigned short*)ws;  ws += 128;                        // zero page

    int n;
    n = B_ * T_ * D_;  cvt_copy<<<(n + 255) / 256, 256, 0, stream>>>(x, xb, n);
    n = KD * H_;       cvt_wT  <<<(n + 255) / 256, 256, 0, stream>>>(f_after, Wc, KD, H_);
    n = D_ * G3H;      cvt_wT  <<<(n + 255) / 256, 256, 0, stream>>>(gru_k,  Wk, D_, G3H);
    n = H_ * G3H;      cvt_wT  <<<(n + 255) / 256, 256, 0, stream>>>(gru_rk, Wr, H_, G3H);
    init_zpad<<<1, 64, 0, stream>>>(zp);

    // conv-as-GEMM, LDS-tiled + double-buffered async-to-LDS pipeline
    conv_gemm_tiled<<<128 * 4, 512, 0, stream>>>(xb, Wc, zp, za);

    {   // xm[t][b][n] = x_t @ gru_kernel  (bias added in scan)
        int waves = T_ * (G3H / 32);
        gemm_kernel<G3H, 1, 0, false><<<(waves * 32) / 256, 256, 0, stream>>>(xb, Wk, xm);
    }
    scan_kernel<<<1, 1024, 0, stream>>>(xm, za, Wr, f_skip, gru_b, ug, out);
}
